// GraphEncoder_5738076307743
// MI455X (gfx1250) — compile-verified, matrix operand order checked by reading, not run on hardware
//
#include <hip/hip_runtime.h>
#include <hip/hip_bf16.h>

typedef _Float16 v16h __attribute__((ext_vector_type(16)));
typedef float    v8f  __attribute__((ext_vector_type(8)));

#define N_NODES 50000
#define N_EDGES 800000
#define F_IN    128
#define HID     64
#define HEADS   4
#define HC      256          // HEADS * HID
#define F_OUT   256
#define NEG_SLOPE 0.2f
#define LN_EPS    1e-5f

// ---------------------------------------------------------------------------
// B prepack: B[K,Nn] f32 row-major -> fragment-major f16.
// Fragment f = kt*tilesN + nt holds the 32x16 (KxN) slice for one WMMA step,
// stored so lane l's 16 halves are contiguous:  Bp[f*512 + l*16 + i].
// Per-lane element map (16-bit B mirror of A 16x32 layout, ISA 7.12.2):
//   lanes 0-15 : col n = l,    K = {0..7, 16..23}   (+ kt*32)
//   lanes 16-31: col n = l-16, K = {8..15, 24..31}  (+ kt*32)
// ---------------------------------------------------------------------------
__global__ void prepack_b_f16(const float* __restrict__ B, _Float16* __restrict__ Bp,
                              int K, int Nn)
{
    int t = blockIdx.x * blockDim.x + threadIdx.x;
    int tilesN = Nn >> 4;
    int frags  = (K >> 5) * tilesN;
    if (t >= frags * 32) return;
    int f = t >> 5, lane = t & 31;
    int kt = f / tilesN, nt = f % tilesN;
    int kb = (lane >> 4) << 3;
    int n  = (nt << 4) + (lane & 15);
    _Float16* dst = Bp + ((size_t)f << 9) + (lane << 4);
#pragma unroll
    for (int i = 0; i < 8; ++i) {
        dst[i]     = (_Float16)B[(size_t)((kt << 5) + kb + i)      * Nn + n];
        dst[i + 8] = (_Float16)B[(size_t)((kt << 5) + 16 + kb + i) * Nn + n];
    }
}

// ---------------------------------------------------------------------------
// WMMA GEMM, 4-way N-blocked: one wave computes a 16x64 output strip
// (4 accumulators share one A fragment -> 4x less A traffic, 4 wmma/k-step).
// C[M,Nn] = A[M,K] (f32 row-major) x B (prepacked f16 fragments).
// M%16==0, Nn%64==0, K%32==0 (true for all GEMMs here; 50000 = 3125*16).
// Optional fused bias + ReLU epilogue. Early-exit is wave-uniform so EXEC
// stays all-1s around the WMMAs as the ISA requires.
// ---------------------------------------------------------------------------
__global__ void wmma_gemm_f32(const float* __restrict__ A,
                              const _Float16* __restrict__ Bp,
                              float* __restrict__ C,
                              int M, int K, int Nn,
                              const float* __restrict__ bias, int relu)
{
    int wave   = (blockIdx.x * blockDim.x + threadIdx.x) >> 5;
    int tilesN = Nn >> 4;
    int nblks  = tilesN >> 2;                  // 4 n-tiles per wave
    if (wave >= (M >> 4) * nblks) return;

    int m0   = (wave / nblks) << 4;
    int nblk =  wave % nblks;
    int lane = threadIdx.x & 31;
    int half = lane >> 4;                      // 0: lanes 0-15, 1: lanes 16-31
    int kb   = half << 3;
    int mrow = m0 + (lane & 15);

    const float* Arow = A + (size_t)mrow * K;
    v8f acc[4] = {};

    int ktiles = K >> 5;
    for (int kt = 0; kt < ktiles; ++kt) {
        int k0 = kt << 5;
        __builtin_prefetch(Arow + k0 + 32, 0, 1);   // next A chunk -> global_prefetch_b8
        v16h a;
#pragma unroll
        for (int i = 0; i < 8; ++i) {               // contiguous per lane -> b128 loads
            a[i]     = (_Float16)Arow[k0 + kb + i];
            a[i + 8] = (_Float16)Arow[k0 + 16 + kb + i];
        }
        const _Float16* bp = Bp + (((size_t)kt * tilesN + (nblk << 2)) << 9) + (lane << 4);
        v16h b0 = *(const v16h*)(bp);
        v16h b1 = *(const v16h*)(bp + 512);
        v16h b2 = *(const v16h*)(bp + 1024);
        v16h b3 = *(const v16h*)(bp + 1536);
        acc[0] = __builtin_amdgcn_wmma_f32_16x16x32_f16(false, a, false, b0, (short)0, acc[0], false, false);
        acc[1] = __builtin_amdgcn_wmma_f32_16x16x32_f16(false, a, false, b1, (short)0, acc[1], false, false);
        acc[2] = __builtin_amdgcn_wmma_f32_16x16x32_f16(false, a, false, b2, (short)0, acc[2], false, false);
        acc[3] = __builtin_amdgcn_wmma_f32_16x16x32_f16(false, a, false, b3, (short)0, acc[3], false, false);
    }

    // C/D layout: VGPR r, lane l -> row m0 + r + half*8, col = n-tile base + (l&15)
    int rbase = m0 + (half << 3);
#pragma unroll
    for (int j = 0; j < 4; ++j) {
        int col  = (nblk << 6) + (j << 4) + (lane & 15);
        float bv = bias ? bias[col] : 0.0f;
#pragma unroll
        for (int r = 0; r < 8; ++r) {
            float v = acc[j][r] + bv;
            if (relu) v = fmaxf(v, 0.0f);
            C[(size_t)(rbase + r) * Nn + col] = v;
        }
    }
}

// ---------------------------------------------------------------------------
// a_src[n,h] = <ht[n,h,:], att_s[h,:]>, a_dst[n,h] = <ht[n,h,:], att_d[h,:]>
// ---------------------------------------------------------------------------
__global__ void att_scores(const float* __restrict__ ht,
                           const float* __restrict__ att_s,
                           const float* __restrict__ att_d,
                           float* __restrict__ as_, float* __restrict__ ad_)
{
    int t = blockIdx.x * blockDim.x + threadIdx.x;
    if (t >= N_NODES * HEADS) return;
    int n = t >> 2, h = t & 3;
    const float* row = ht + (size_t)n * HC + h * HID;
    const float* ws  = att_s + h * HID;
    const float* wd  = att_d + h * HID;
    float ss = 0.f, sd = 0.f;
#pragma unroll 8
    for (int c = 0; c < HID; ++c) { float v = row[c]; ss += v * ws[c]; sd += v * wd[c]; }
    as_[t] = ss;
    ad_[t] = sd;
}

// order-preserving float <-> uint map for atomicMax-based segment max
__device__ __forceinline__ unsigned fenc(float x) {
    unsigned b = __float_as_uint(x);
    return (b & 0x80000000u) ? ~b : (b | 0x80000000u);
}
__device__ __forceinline__ float fdec(unsigned u) {
    return (u & 0x80000000u) ? __uint_as_float(u ^ 0x80000000u) : __uint_as_float(~u);
}

__device__ __forceinline__ void edge_ids(int i, const int* __restrict__ esrc,
                                         const int* __restrict__ edst, int& s, int& d) {
    if (i < N_EDGES) { s = esrc[i]; d = edst[i]; }
    else             { s = i - N_EDGES; d = s; }      // self loops appended
}

__global__ void edge_max(const int* __restrict__ esrc, const int* __restrict__ edst,
                         const float* __restrict__ as_, const float* __restrict__ ad_,
                         unsigned* __restrict__ menc)
{
    int i = blockIdx.x * blockDim.x + threadIdx.x;
    if (i >= N_EDGES + N_NODES) return;
    int s, d; edge_ids(i, esrc, edst, s, d);
#pragma unroll
    for (int h = 0; h < HEADS; ++h) {
        float e = as_[s * HEADS + h] + ad_[d * HEADS + h];
        e = (e > 0.f) ? e : NEG_SLOPE * e;
        atomicMax(&menc[d * HEADS + h], fenc(e));
    }
}

__global__ void edge_exp(const int* __restrict__ esrc, const int* __restrict__ edst,
                         const float* __restrict__ as_, const float* __restrict__ ad_,
                         const unsigned* __restrict__ menc,
                         float* __restrict__ exs, float* __restrict__ zsum)
{
    int i = blockIdx.x * blockDim.x + threadIdx.x;
    if (i >= N_EDGES + N_NODES) return;
    int s, d; edge_ids(i, esrc, edst, s, d);
#pragma unroll
    for (int h = 0; h < HEADS; ++h) {
        float e = as_[s * HEADS + h] + ad_[d * HEADS + h];
        e = (e > 0.f) ? e : NEG_SLOPE * e;
        float ex = __expf(e - fdec(menc[d * HEADS + h]));
        exs[(size_t)i * HEADS + h] = ex;
        atomicAdd(&zsum[d * HEADS + h], ex);
    }
}

// one block per edge, one lane per channel: agg[dst,c] += ht[src,c] * alpha[h(c)]
// Bound by L2 atomics; all per-node state (~154 MB) fits the 192 MB L2.
__global__ void edge_scatter(const int* __restrict__ esrc, const int* __restrict__ edst,
                             const float* __restrict__ ht,
                             const float* __restrict__ exs, const float* __restrict__ zsum,
                             float* __restrict__ agg)
{
    int i = blockIdx.x;
    int c = threadIdx.x;                 // 0..255
    int s, d; edge_ids(i, esrc, edst, s, d);
    int h = c >> 6;
    float alpha = exs[(size_t)i * HEADS + h] / (zsum[d * HEADS + h] + 1e-16f);
    atomicAdd(&agg[(size_t)d * HC + c], ht[(size_t)s * HC + c] * alpha);
}

// per-node: feat = relu(layernorm(agg + bias) * g + b)
__global__ void bias_ln_relu(const float* __restrict__ agg, const float* __restrict__ bias,
                             const float* __restrict__ g, const float* __restrict__ b,
                             float* __restrict__ outp)
{
    __shared__ float sh[HC];
    int n = blockIdx.x, c = threadIdx.x;
    float v = agg[(size_t)n * HC + c] + bias[c];
    sh[c] = v; __syncthreads();
    for (int s = HC / 2; s > 0; s >>= 1) { if (c < s) sh[c] += sh[c + s]; __syncthreads(); }
    float mu = sh[0] * (1.0f / HC);
    __syncthreads();
    float dv = v - mu;
    sh[c] = dv * dv; __syncthreads();
    for (int s = HC / 2; s > 0; s >>= 1) { if (c < s) sh[c] += sh[c + s]; __syncthreads(); }
    float rs = rsqrtf(sh[0] * (1.0f / HC) + LN_EPS);
    outp[(size_t)n * HC + c] = fmaxf(dv * rs * g[c] + b[c], 0.0f);
}

// mean pool stage 1: per-block partial column sums -> atomic add into gsum[256]
__global__ void pool_sum(const float* __restrict__ feat, float* __restrict__ gsum)
{
    int c  = threadIdx.x;
    int r0 = blockIdx.x * 128;
    int r1 = min(r0 + 128, N_NODES);
    float acc = 0.f;
    for (int r = r0; r < r1; ++r) acc += feat[(size_t)r * HC + c];
    atomicAdd(&gsum[c], acc);
}

// out[o] = (gsum/N) @ w_out + b_out
__global__ void out_proj(const float* __restrict__ gsum, const float* __restrict__ w_out,
                         const float* __restrict__ b_out, float* __restrict__ outp)
{
    int o = threadIdx.x;                 // 256
    const float invN = 1.0f / (float)N_NODES;
    float acc = 0.f;
    for (int c = 0; c < HC; ++c) acc += (gsum[c] * invN) * w_out[c * F_OUT + o];
    outp[o] = acc + b_out[o];
}

static inline void run_gemm(const float* A, const float* B, _Float16* Bp, float* C,
                            int M, int K, int Nn, const float* bias, int relu,
                            hipStream_t stream)
{
    int frags = (K >> 5) * (Nn >> 4);
    prepack_b_f16<<<(frags * 32 + 255) / 256, 256, 0, stream>>>(B, Bp, K, Nn);
    int waves = (M >> 4) * (Nn >> 6);            // 4 n-tiles per wave
    wmma_gemm_f32<<<(waves + 7) / 8, 256, 0, stream>>>(A, Bp, C, M, K, Nn, bias, relu);
}

extern "C" void kernel_launch(void* const* d_in, const int* in_sizes, int n_in,
                              void* d_out, int out_size, void* d_ws, size_t ws_size,
                              hipStream_t stream)
{
    const float* x      = (const float*)d_in[0];
    const int*   ei     = (const int*)  d_in[1];
    const float* w_in   = (const float*)d_in[2];
    const float* b_in   = (const float*)d_in[3];
    const float* lin0_w = (const float*)d_in[4];
    const float* att_s0 = (const float*)d_in[5];
    const float* att_d0 = (const float*)d_in[6];
    const float* bias0  = (const float*)d_in[7];
    const float* ln0_g  = (const float*)d_in[8];
    const float* ln0_b  = (const float*)d_in[9];
    const float* lin1_w = (const float*)d_in[10];
    const float* att_s1 = (const float*)d_in[11];
    const float* att_d1 = (const float*)d_in[12];
    const float* bias1  = (const float*)d_in[13];
    const float* ln1_g  = (const float*)d_in[14];
    const float* ln1_b  = (const float*)d_in[15];
    const float* w_out  = (const float*)d_in[16];
    const float* b_out  = (const float*)d_in[17];
    const int* esrc = ei;
    const int* edst = ei + N_EDGES;

    // workspace carve-up (floats unless noted); ~183 MB total
    float* h0      = (float*)d_ws;                               // [N,64]
    float* ht      = h0   + (size_t)N_NODES * HID;               // [N,256]
    float* agg     = ht   + (size_t)N_NODES * HC;                // [N,256]
    float* feat    = agg  + (size_t)N_NODES * HC;                // [N,256]
    float* as_     = feat + (size_t)N_NODES * HC;                // [N,4]
    float* ad_     = as_  + (size_t)N_NODES * HEADS;             // [N,4]
    unsigned* menc = (unsigned*)(ad_ + (size_t)N_NODES * HEADS); // [N,4]
    float* zsum    = (float*)(menc + (size_t)N_NODES * HEADS);   // [N,4]
    float* exs     = zsum + (size_t)N_NODES * HEADS;             // [E+N,4]
    float* gsum    = exs  + (size_t)(N_EDGES + N_NODES) * HEADS; // [256]
    _Float16* Bp   = (_Float16*)(gsum + 256);                    // <=128 KB, 64B-aligned

    const int etot = N_EDGES + N_NODES;
    dim3 blk(256);

    // h0 = relu(x @ w_in + b_in)
    run_gemm(x, w_in, Bp, h0, N_NODES, F_IN, HID, b_in, 1, stream);

    for (int layer = 0; layer < 2; ++layer) {
        const float* fin  = layer ? feat   : h0;
        int          fk   = layer ? HC     : HID;
        const float* linw = layer ? lin1_w : lin0_w;
        const float* ats  = layer ? att_s1 : att_s0;
        const float* atd  = layer ? att_d1 : att_d0;
        const float* bs   = layer ? bias1  : bias0;
        const float* lg   = layer ? ln1_g  : ln0_g;
        const float* lb   = layer ? ln1_b  : ln0_b;

        // ht = fin @ lin_w   (WMMA)
        run_gemm(fin, linw, Bp, ht, N_NODES, fk, HC, nullptr, 0, stream);

        att_scores<<<(N_NODES * HEADS + 255) / 256, blk, 0, stream>>>(ht, ats, atd, as_, ad_);

        hipMemsetAsync(menc, 0, (size_t)N_NODES * HEADS * sizeof(unsigned), stream);
        hipMemsetAsync(zsum, 0, (size_t)N_NODES * HEADS * sizeof(float),    stream);
        hipMemsetAsync(agg,  0, (size_t)N_NODES * HC    * sizeof(float),    stream);

        edge_max    <<<(etot + 255) / 256, blk, 0, stream>>>(esrc, edst, as_, ad_, menc);
        edge_exp    <<<(etot + 255) / 256, blk, 0, stream>>>(esrc, edst, as_, ad_, menc, exs, zsum);
        edge_scatter<<<etot,              blk, 0, stream>>>(esrc, edst, ht, exs, zsum, agg);

        bias_ln_relu<<<N_NODES, blk, 0, stream>>>(agg, bs, lg, lb, feat);
    }

    hipMemsetAsync(gsum, 0, HC * sizeof(float), stream);
    pool_sum<<<(N_NODES + 127) / 128, blk, 0, stream>>>(feat, gsum);
    out_proj<<<1, blk, 0, stream>>>(gsum, w_out, b_out, (float*)d_out);
}
